// CalderaLinear_22385369547107
// MI455X (gfx1250) — matrix operand order repaired and checked by reading, not compiled, not run on hardware
//
#include <hip/hip_runtime.h>
#include <hip/hip_bf16.h>

// ---------------------------------------------------------------------------
// CalderaLinear for MI455X (gfx1250): LDS-staged bf16 WMMA GEMM with exact
// per-group fp32 scaling. Tile staging via Tensor Data Mover (TDM) with
// double buffering; falls back to async global->LDS, then sync loads.
// out = x @ deq(q).T + (x @ deq(r).T) @ deq(l).T + bias
// ---------------------------------------------------------------------------

typedef __attribute__((ext_vector_type(16))) __bf16 v16bf;
typedef __attribute__((ext_vector_type(8)))  __bf16 v8bf;
typedef __attribute__((ext_vector_type(4)))  __bf16 v4bf;
typedef __attribute__((ext_vector_type(8)))  float  v8f;

typedef __attribute__((ext_vector_type(4))) unsigned int u32x4;
typedef __attribute__((ext_vector_type(8))) int i32x8;
typedef __attribute__((ext_vector_type(4))) int i32x4;

// Exact pointee type expected by __builtin_amdgcn_global_load_async_to_lds_b128
typedef int async_v4i __attribute__((vector_size(4 * sizeof(int))));

#define N_ROWS 8192   // B*S = 4*2048
#define D_IN   4096
#define D_OUT  4096
#define RANK   256
#define GROUPK 128    // quant group size along K

#define BM    128     // block tile M
#define BN    128     // block tile N
#define BK    128     // staged K per group (== GROUPK)
#define LDSP  136     // padded LDS row stride (elems): 272B -> conflict-free b128
#define TILE  (BM * LDSP)

#define CONCAT16(lo, hi) \
  __builtin_shufflevector(lo, hi, 0,1,2,3,4,5,6,7,8,9,10,11,12,13,14,15)

#if __has_builtin(__builtin_amdgcn_tensor_load_to_lds) && \
    __has_builtin(__builtin_amdgcn_s_wait_tensorcnt)
#define HAVE_TDM 1
#else
#define HAVE_TDM 0
#endif

#if __has_builtin(__builtin_amdgcn_global_load_async_to_lds_b128) && \
    __has_builtin(__builtin_amdgcn_s_wait_asynccnt)
#define HAVE_ASYNC 1
#else
#define HAVE_ASYNC 0
#endif

#if HAVE_TDM || HAVE_ASYNC
#define NBUF 2
#else
#define NBUF 1
#endif

// ---------------------------------------------------------------------------
// Elementwise converts (4-wide)
// ---------------------------------------------------------------------------
__global__ __launch_bounds__(256)
void cvt_f32_bf16_v4(const float* __restrict__ in, __bf16* __restrict__ out, int n4) {
  int i = blockIdx.x * blockDim.x + threadIdx.x;
  if (i >= n4) return;
  const float4 v = reinterpret_cast<const float4*>(in)[i];
  v4bf o;
  o[0] = (__bf16)v.x; o[1] = (__bf16)v.y; o[2] = (__bf16)v.z; o[3] = (__bf16)v.w;
  reinterpret_cast<v4bf*>(out)[i] = o;
}

__global__ __launch_bounds__(256)
void cvt_i32_bf16_v4(const int* __restrict__ in, __bf16* __restrict__ out, int n4) {
  int i = blockIdx.x * blockDim.x + threadIdx.x;
  if (i >= n4) return;
  const int4 v = reinterpret_cast<const int4*>(in)[i];
  v4bf o;  // values are 0..15 -> exact in bf16
  o[0] = (__bf16)(float)v.x; o[1] = (__bf16)(float)v.y;
  o[2] = (__bf16)(float)v.z; o[3] = (__bf16)(float)v.w;
  reinterpret_cast<v4bf*>(out)[i] = o;
}

#if HAVE_TDM
// ---------------------------------------------------------------------------
// TDM: DMA one 128-row x 128-elem (bf16) 2-D tile from global into LDS.
// D# per CDNA5 ISA ch.8: group0 {count, lds_addr, global_addr, type=2},
// group1 {data_size=2B, pad 16B after every 256B row (-> 272B LDS stride),
//         tensor_dim0=ld, tensor_dim1=128, tile 128x128, dim0_stride=ld}.
// Issued by one wave; completion tracked with TENSORcnt.
// ---------------------------------------------------------------------------
__device__ __forceinline__ void tdm_stage_tile(
    const __bf16* gsrc, int ld, unsigned lds_byte_addr) {
  const unsigned long long ga = (unsigned long long)gsrc;
  u32x4 g0;
  g0[0] = 1u;                                   // count=1, user descriptor
  g0[1] = lds_byte_addr;                        // lds_addr [63:32]
  g0[2] = (unsigned)(ga & 0xFFFFFFFFu);         // global_addr lo
  g0[3] = (unsigned)((ga >> 32) & 0x1FFFFFFu)   // global_addr hi (57-bit)
        | (2u << 30);                           // type=2 ("image")
  i32x8 g1;
  g1[0] = (int)((1u << 16)      // data_size: 2 bytes
              | (1u << 20)      // pad_enable
              | (5u << 22)      // pad_interval: 2^5 * 8B = 256B
              | (3u << 25));    // pad_amount: 4 DWORDs = 16B
  g1[1] = (int)(((unsigned)ld & 0xFFFFu) << 16);          // tensor_dim0[15:0]
  g1[2] = (int)(((unsigned)ld >> 16) | (128u << 16));     // dim0[31:16] | tensor_dim1=128
  g1[3] = (int)(128u << 16);                              // tile_dim0=128
  g1[4] = (int)128u;                                      // tile_dim1=128, tile_dim2=0
  g1[5] = ld;                                             // tensor_dim0_stride lo
  g1[6] = 0;                                              // stride hi | dim1_stride lo
  g1[7] = 0;
  i32x4 gz = {};
#if __clang_major__ >= 23
  i32x8 gz8 = {};
  __builtin_amdgcn_tensor_load_to_lds(g0, g1, gz, gz, gz8, 0);
#else
  __builtin_amdgcn_tensor_load_to_lds(g0, g1, gz, gz, 0);
#endif
}
#endif  // HAVE_TDM

// ---------------------------------------------------------------------------
// Cooperative staging (fallback tiers): 2048 16B-chunks/tile, 8 per thread.
// ---------------------------------------------------------------------------
__device__ __forceinline__ void stage_group(
    const __bf16* __restrict__ A, int lda, int m_blk,
    const __bf16* __restrict__ W, int ldw, int n_blk,
    int g, __bf16* dA, __bf16* dB, int tid) {
  const int k = g * BK;
#pragma unroll
  for (int i = 0; i < (BM * BK / 8) / 256; ++i) {   // 8 iterations
    const int c   = tid + i * 256;
    const int row = c >> 4;
    const int cc  = (c & 15) * 8;
    const __bf16* gA = A + (size_t)(m_blk + row) * lda + k + cc;
    const __bf16* gB = W + (size_t)(n_blk + row) * ldw + k + cc;
    __bf16* lA = dA + row * LDSP + cc;
    __bf16* lB = dB + row * LDSP + cc;
#if HAVE_ASYNC
    __builtin_amdgcn_global_load_async_to_lds_b128(
        (__attribute__((address_space(1))) async_v4i*)gA,
        (__attribute__((address_space(3))) async_v4i*)lA, 0, 0);
    __builtin_amdgcn_global_load_async_to_lds_b128(
        (__attribute__((address_space(1))) async_v4i*)gB,
        (__attribute__((address_space(3))) async_v4i*)lB, 0, 0);
#else
    *(v8bf*)lA = *(const v8bf*)gA;
    *(v8bf*)lB = *(const v8bf*)gB;
#endif
  }
}

// ---------------------------------------------------------------------------
// One K-group of WMMA from LDS. Wave tile: 32(M) x 64(N) -> 2x4 accumulators.
// ---------------------------------------------------------------------------
__device__ __forceinline__ void compute_group(
    const __bf16* sA, const __bf16* sB,
    int m_off, int n_off, int l15, int half, v8f (&accg)[2][4]) {
#pragma unroll
  for (int kk = 0; kk < 4; ++kk) {
    v16bf af[2];
#pragma unroll
    for (int mt = 0; mt < 2; ++mt) {
      const __bf16* pa = sA + (m_off + mt * 16 + l15) * LDSP + kk * 32 + half * 8;
      v8bf lo = *(const v8bf*)pa;
      v8bf hi = *(const v8bf*)(pa + 16);
      af[mt] = CONCAT16(lo, hi);
    }
#pragma unroll
    for (int nt = 0; nt < 4; ++nt) {
      const __bf16* pb = sB + (n_off + nt * 16 + l15) * LDSP + kk * 32 + half * 8;
      v8bf lo = *(const v8bf*)pb;
      v8bf hi = *(const v8bf*)(pb + 16);
      v16bf bfrag = CONCAT16(lo, hi);
#pragma unroll
      for (int mt = 0; mt < 2; ++mt)
        accg[mt][nt] = __builtin_amdgcn_wmma_f32_16x16x32_bf16(
            false, af[mt], false, bfrag, (short)0, accg[mt][nt], false, false);
    }
  }
}

// Group accumulate + exact fp32 group scaling.
__device__ __forceinline__ void group_accumulate(
    const __bf16* sA, const __bf16* sB, int g,
    const float* __restrict__ S, int sstride, int n_blk,
    int m_off, int n_off, int l15, int half, v8f (&accm)[2][4]) {
  v8f accg[2][4];
#pragma unroll
  for (int mt = 0; mt < 2; ++mt)
#pragma unroll
    for (int nt = 0; nt < 4; ++nt)
#pragma unroll
      for (int i = 0; i < 8; ++i) accg[mt][nt][i] = 0.0f;

  compute_group(sA, sB, m_off, n_off, l15, half, accg);

#pragma unroll
  for (int nt = 0; nt < 4; ++nt) {
    const float s = S[(size_t)(n_blk + n_off + nt * 16 + l15) * sstride + g];
#pragma unroll
    for (int mt = 0; mt < 2; ++mt) accm[mt][nt] += accg[mt][nt] * s;
  }
}

// ---------------------------------------------------------------------------
// Full K-phase, double-buffered through LDS.
// ---------------------------------------------------------------------------
__device__ __forceinline__ void run_phase_lds(
    const __bf16* __restrict__ A, int lda, int m_blk,
    const __bf16* __restrict__ W, int ldw, int n_blk,
    const float* __restrict__ S, int sstride, int ngroups,
    __bf16* smA, __bf16* smB, v8f (&accm)[2][4]) {
  const int tid  = threadIdx.x;
  const int lane = tid & 31, w = tid >> 5;
  const int wm = w >> 1, wn = w & 1;
  const int l15 = lane & 15, half = lane >> 4;
  const int m_off = wm * 32, n_off = wn * 64;

#if HAVE_TDM
  const unsigned ldsA_b =
      (unsigned)(unsigned long long)(__attribute__((address_space(3))) char*)smA;
  const unsigned ldsB_b =
      (unsigned)(unsigned long long)(__attribute__((address_space(3))) char*)smB;
  if (w == 0) {
    tdm_stage_tile(A + (size_t)m_blk * lda, lda, ldsA_b);
    tdm_stage_tile(W + (size_t)n_blk * ldw, ldw, ldsB_b);
  }
  for (int g = 0; g < ngroups; ++g) {
    if (w == 0) __builtin_amdgcn_s_wait_tensorcnt(0);
    __syncthreads();
    if (w == 0 && g + 1 < ngroups) {
      const unsigned boff = (unsigned)(((g + 1) & 1) * TILE * 2);
      tdm_stage_tile(A + (size_t)m_blk * lda + (g + 1) * BK, lda, ldsA_b + boff);
      tdm_stage_tile(W + (size_t)n_blk * ldw + (g + 1) * BK, ldw, ldsB_b + boff);
    }
    group_accumulate(smA + (g & 1) * TILE, smB + (g & 1) * TILE, g,
                     S, sstride, n_blk, m_off, n_off, l15, half, accm);
  }
#elif HAVE_ASYNC
  stage_group(A, lda, m_blk, W, ldw, n_blk, 0, smA, smB, tid);
  for (int g = 0; g < ngroups; ++g) {
    __builtin_amdgcn_s_wait_asynccnt(0);
    __syncthreads();
    if (g + 1 < ngroups)
      stage_group(A, lda, m_blk, W, ldw, n_blk, g + 1,
                  smA + ((g + 1) & 1) * TILE, smB + ((g + 1) & 1) * TILE, tid);
    group_accumulate(smA + (g & 1) * TILE, smB + (g & 1) * TILE, g,
                     S, sstride, n_blk, m_off, n_off, l15, half, accm);
  }
#else
  for (int g = 0; g < ngroups; ++g) {
    __syncthreads();
    stage_group(A, lda, m_blk, W, ldw, n_blk, g, smA, smB, tid);
    __syncthreads();
    group_accumulate(smA, smB, g, S, sstride, n_blk,
                     m_off, n_off, l15, half, accm);
  }
#endif
}

// ---------------------------------------------------------------------------
// xr = x @ deq(r).T -> bf16 [N_ROWS, RANK].  Block tile 128M x 128N.
// ---------------------------------------------------------------------------
__global__ __launch_bounds__(256)
void caldera_xr(const __bf16* __restrict__ xbf, const __bf16* __restrict__ rbf,
                const float* __restrict__ r_scales, __bf16* __restrict__ xr) {
  __shared__ __bf16 smA[NBUF * TILE];
  __shared__ __bf16 smB[NBUF * TILE];
  const int lane = threadIdx.x & 31, w = threadIdx.x >> 5;
  const int wm = w >> 1, wn = w & 1;
  const int l15 = lane & 15, half = lane >> 4;
  const int m_blk = blockIdx.y * BM;
  const int n_blk = blockIdx.x * BN;

  v8f accm[2][4];
#pragma unroll
  for (int mt = 0; mt < 2; ++mt)
#pragma unroll
    for (int nt = 0; nt < 4; ++nt)
#pragma unroll
      for (int i = 0; i < 8; ++i) accm[mt][nt][i] = 0.0f;

  run_phase_lds(xbf, D_IN, m_blk, rbf, D_IN, n_blk,
                r_scales, D_IN / GROUPK, D_IN / GROUPK, smA, smB, accm);

#pragma unroll
  for (int mt = 0; mt < 2; ++mt)
#pragma unroll
    for (int nt = 0; nt < 4; ++nt) {
      const int n = n_blk + wn * 64 + nt * 16 + l15;
#pragma unroll
      for (int i = 0; i < 8; ++i) {
        const int m = m_blk + wm * 32 + mt * 16 + half * 8 + i;
        xr[(size_t)m * RANK + n] = (__bf16)accm[mt][nt][i];
      }
    }
}

// ---------------------------------------------------------------------------
// out = x @ deq(q).T + xr @ deq(l).T + bias  (virtual K = 4096 + 256)
// ---------------------------------------------------------------------------
__global__ __launch_bounds__(256)
void caldera_main(const __bf16* __restrict__ xbf, const __bf16* __restrict__ qbf,
                  const float* __restrict__ q_scales,
                  const __bf16* __restrict__ xrbf, const __bf16* __restrict__ lbf,
                  const float* __restrict__ l_scales,
                  const float* __restrict__ bias, float* __restrict__ out) {
  __shared__ __bf16 smA[NBUF * TILE];
  __shared__ __bf16 smB[NBUF * TILE];
  const int lane = threadIdx.x & 31, w = threadIdx.x >> 5;
  const int wm = w >> 1, wn = w & 1;
  const int l15 = lane & 15, half = lane >> 4;
  const int m_blk = blockIdx.y * BM;
  const int n_blk = blockIdx.x * BN;

  v8f accm[2][4];
#pragma unroll
  for (int mt = 0; mt < 2; ++mt)
#pragma unroll
    for (int nt = 0; nt < 4; ++nt)
#pragma unroll
      for (int i = 0; i < 8; ++i) accm[mt][nt][i] = 0.0f;

  // Phase 1: dense quantized weight, 32 groups over K=4096
  run_phase_lds(xbf, D_IN, m_blk, qbf, D_IN, n_blk,
                q_scales, D_IN / GROUPK, D_IN / GROUPK, smA, smB, accm);
  // Phase 2: low-rank correction, 2 groups over K=256
  run_phase_lds(xrbf, RANK, m_blk, lbf, RANK, n_blk,
                l_scales, RANK / GROUPK, RANK / GROUPK, smA, smB, accm);

#pragma unroll
  for (int mt = 0; mt < 2; ++mt)
#pragma unroll
    for (int nt = 0; nt < 4; ++nt) {
      const int n = n_blk + wn * 64 + nt * 16 + l15;
      const float b = bias[n];
#pragma unroll
      for (int i = 0; i < 8; ++i) {
        const int m = m_blk + wm * 32 + mt * 16 + half * 8 + i;
        out[(size_t)m * D_OUT + n] = accm[mt][nt][i] + b;
      }
    }
}

// ---------------------------------------------------------------------------
// Launch
// ---------------------------------------------------------------------------
extern "C" void kernel_launch(void* const* d_in, const int* in_sizes, int n_in,
                              void* d_out, int out_size, void* d_ws, size_t ws_size,
                              hipStream_t stream) {
  (void)in_sizes; (void)n_in; (void)out_size; (void)ws_size;

  const float* x        = (const float*)d_in[0];
  const int*   q_vals   = (const int*)  d_in[1];
  const float* q_scales = (const float*)d_in[2];
  const int*   l_vals   = (const int*)  d_in[3];
  const float* l_scales = (const float*)d_in[4];
  const int*   r_vals   = (const int*)  d_in[5];
  const float* r_scales = (const float*)d_in[6];
  const float* bias     = (const float*)d_in[7];
  float* out = (float*)d_out;

  // Workspace layout (bf16): x | q | l | r | xr   (~109 MB total)
  char* ws = (char*)d_ws;
  __bf16* xbf  = (__bf16*)ws; ws += (size_t)N_ROWS * D_IN * sizeof(__bf16);
  __bf16* qbf  = (__bf16*)ws; ws += (size_t)D_OUT  * D_IN * sizeof(__bf16);
  __bf16* lbf  = (__bf16*)ws; ws += (size_t)D_OUT  * RANK * sizeof(__bf16);
  __bf16* rbf  = (__bf16*)ws; ws += (size_t)RANK   * D_IN * sizeof(__bf16);
  __bf16* xrbf = (__bf16*)ws;

  {
    int n4 = N_ROWS * D_IN / 4;
    cvt_f32_bf16_v4<<<n4 / 256, 256, 0, stream>>>(x, xbf, n4);
  }
  {
    int n4 = D_OUT * D_IN / 4;
    cvt_i32_bf16_v4<<<n4 / 256, 256, 0, stream>>>(q_vals, qbf, n4);
  }
  {
    int n4 = D_OUT * RANK / 4;
    cvt_i32_bf16_v4<<<n4 / 256, 256, 0, stream>>>(l_vals, lbf, n4);
  }
  {
    int n4 = RANK * D_IN / 4;
    cvt_i32_bf16_v4<<<n4 / 256, 256, 0, stream>>>(r_vals, rbf, n4);
  }

  // xr = x @ deq(r).T : 2 N-blocks x 64 M-blocks
  {
    dim3 grid(RANK / BN, N_ROWS / BM);
    caldera_xr<<<grid, 256, 0, stream>>>(xbf, rbf, r_scales, xrbf);
  }

  // out = x @ deq(q).T + xr @ deq(l).T + bias
  {
    dim3 grid(D_OUT / BN, N_ROWS / BM);
    caldera_main<<<grid, 256, 0, stream>>>(xbf, qbf, q_scales, xrbf, lbf,
                                           l_scales, bias, out);
  }
}
